// AttentionModel_23304492548756
// MI455X (gfx1250) — matrix operand order repaired and checked by reading, not compiled, not run on hardware
//
#include <hip/hip_runtime.h>
#include <hip/hip_bf16.h>

#define B_  16
#define C_  512
#define S_  512
#define H_  8
#define L_  6
#define FF_ 2048
#define O_  512
#define DH_ 64
#define M_  (B_ * S_)   // 8192 token rows

typedef __attribute__((ext_vector_type(16))) __bf16 v16bf;
typedef __attribute__((ext_vector_type(8)))  float  v8f;

__device__ __forceinline__ unsigned short f2bf(float f) {
  unsigned int u = __float_as_uint(f);
  u += 0x7FFFu + ((u >> 16) & 1u);   // round-to-nearest-even
  return (unsigned short)(u >> 16);
}

union FragU { v16bf v; uint4 u[2]; };

// global -> LDS direct async copy (16B per lane), tracked by ASYNCcnt.
__device__ __forceinline__ void async_copy_b128(unsigned lds_off, const void* gaddr) {
  asm volatile("global_load_async_to_lds_b128 %0, %1, off"
               :: "v"(lds_off), "v"(gaddr) : "memory");
}
__device__ __forceinline__ void wait_async0() {
  asm volatile("s_wait_asynccnt 0x0" ::: "memory");
}

// ---------------------------------------------------------------------------
// bf16 WMMA GEMM: C[M,N] = A[M,K] * Bt[N,K]^T (+bias, +relu) with f32 accum.
// B is ALWAYS supplied transposed: element (k,n) = Bm[n*ldb + k], so both
// A and B staging are contiguous 16B async copies.
// Double-buffered LDS + GLOBAL_LOAD_ASYNC_TO_LDS_B128: tile k0+64 streams
// into buf^1 (no VGPR round-trip, no loadcnt stalls) while WMMAs consume
// tile k0 from buf; one s_wait_asynccnt 0 + one barrier per K-tile.
// Block: 256 threads (8 wave32), tile 128(M) x 64(N), K staged 64 at a time.
// EPI: 0 = store f32; 1 = store bf16; 2 = store bf16 in Vt[b,h,d,s] layout;
//      3 = bias+relu, store bf16; 4 = bias, store f32.
// Batch via blockIdx.z: per-matrix offset = (z/Hdiv)*s?1 + (z%Hdiv)*s?2.
// ---------------------------------------------------------------------------
template <int EPI>
__global__ __launch_bounds__(256, 1) void gemm_bf16_wmma(
    const unsigned short* __restrict__ A, const unsigned short* __restrict__ Bm,
    void* __restrict__ Cout, const float* __restrict__ bias,
    int Mm, int Nn, int Kk, int lda, int ldb, int ldc,
    int Hdiv, long long sA1, long long sA2, long long sB1, long long sB2,
    long long sC1, long long sC2)
{
  __shared__ __align__(16) unsigned short As[2][128][72];  // +8 pad, 2 buffers
  __shared__ __align__(16) unsigned short Bs[2][64][72];   // Bs[n][k]

  const int z  = blockIdx.z;
  const int zb = z / Hdiv, zh = z % Hdiv;
  A  += zb * sA1 + zh * sA2;
  Bm += zb * sB1 + zh * sB2;
  const long long coff = zb * sC1 + zh * sC2;

  const int m0 = blockIdx.y * 128;
  const int n0 = blockIdx.x * 64;
  const int t     = threadIdx.x;
  const int wave  = t >> 5;
  const int lane  = t & 31;
  const int l16   = lane & 15;
  const int lhalf = lane >> 4;
  const int waveM = wave >> 1;   // 0..3
  const int waveN = wave & 1;    // 0..1

  v8f acc[2][2] = {};

  // issue async copies of one 128x64 A tile + 64x64 B^T tile into buffer `buf`
  auto issue_tile = [&](int k0, int buf) {
    #pragma unroll
    for (int r = 0; r < 4; ++r) {
      int slot = t + 256 * r;            // 0..1023
      int row  = slot >> 3;              // 0..127
      int col  = (slot & 7) << 3;        // 0..56
      async_copy_b128((unsigned)(uintptr_t)&As[buf][row][col],
                      &A[(long long)(m0 + row) * lda + k0 + col]);
    }
    #pragma unroll
    for (int r = 0; r < 2; ++r) {
      int slot = t + 256 * r;            // 0..511
      int row  = slot >> 3;              // n: 0..63
      int col  = (slot & 7) << 3;        // k: 0..56
      async_copy_b128((unsigned)(uintptr_t)&Bs[buf][row][col],
                      &Bm[(long long)(n0 + row) * ldb + k0 + col]);
    }
  };

  issue_tile(0, 0);
  int buf = 0;
  for (int k0 = 0; k0 < Kk; k0 += 64) {
    wait_async0();                       // drain this tile's async copies
    __syncthreads();
    if (k0 + 64 < Kk) issue_tile(k0 + 64, buf ^ 1);  // streams during WMMAs

    // ---- two 16x16x32 wmma K-steps on the staged tile ----
    #pragma unroll
    for (int ks = 0; ks < 64; ks += 32) {
      FragU a[2], b[2];
      #pragma unroll
      for (int mi = 0; mi < 2; ++mi) {
        int r = waveM * 32 + mi * 16 + l16;
        // A layout (ISA 7.12.2): lane<16 -> K {0..7,16..23}; lane>=16 -> +8
        a[mi].u[0] = *(const uint4*)&As[buf][r][ks + lhalf * 8];
        a[mi].u[1] = *(const uint4*)&As[buf][r][ks + lhalf * 8 + 16];
      }
      #pragma unroll
      for (int ni = 0; ni < 2; ++ni) {
        int n = waveN * 32 + ni * 16 + l16;
        // B layout: lane<16 -> K {0..7,16..23}; lane>=16 -> +8 (from Bs[n][k])
        b[ni].u[0] = *(const uint4*)&Bs[buf][n][ks + lhalf * 8];
        b[ni].u[1] = *(const uint4*)&Bs[buf][n][ks + lhalf * 8 + 16];
      }
      #pragma unroll
      for (int mi = 0; mi < 2; ++mi)
        #pragma unroll
        for (int ni = 0; ni < 2; ++ni)
          acc[mi][ni] = __builtin_amdgcn_wmma_f32_16x16x32_bf16(
              false, a[mi].v, false, b[ni].v, (short)0, acc[mi][ni], false, false);
    }
    buf ^= 1;
  }

  // ---- epilogue: C/D layout lane -> (m = vgpr + 8*lanehalf, n = lane%16) ----
  #pragma unroll
  for (int mi = 0; mi < 2; ++mi) {
    #pragma unroll
    for (int ni = 0; ni < 2; ++ni) {
      int ng = n0 + waveN * 32 + ni * 16 + l16;
      float bval = (EPI >= 3) ? bias[ng] : 0.0f;
      #pragma unroll
      for (int v = 0; v < 8; ++v) {
        int mg = m0 + waveM * 32 + mi * 16 + lhalf * 8 + v;
        float val = acc[mi][ni][v] + bval;
        if (EPI == 3) val = fmaxf(val, 0.0f);
        if (EPI == 0 || EPI == 4) {
          ((float*)Cout)[coff + (long long)mg * ldc + ng] = val;
        } else if (EPI == 2) {
          // Vt[b, h, d, s]; mg = b*512 + s, ng = h*64 + d (contiguous in s)
          long long idx = (long long)(mg >> 9) * (C_ * S_)
                        + (long long)(ng >> 6) * (DH_ * S_)
                        + (long long)(ng & 63) * S_ + (mg & 511);
          ((unsigned short*)Cout)[idx] = f2bf(val);
        } else {
          ((unsigned short*)Cout)[coff + (long long)mg * ldc + ng] = f2bf(val);
        }
      }
    }
  }
}

// ---------------------------------------------------------------------------
// Weight conversion: f32 [K,N] row-major -> bf16 [N,K] row-major (transposed),
// LDS-tiled 32x32 so both global sides stay coalesced. grid.z = matrix index.
// ---------------------------------------------------------------------------
__global__ __launch_bounds__(256) void transpose_bf16_k(
    const float* __restrict__ in, unsigned short* __restrict__ out, int K, int N) {
  __shared__ unsigned short tile[32][33];
  const float* inm = in + (long long)blockIdx.z * K * N;
  unsigned short* outm = out + (long long)blockIdx.z * K * N;
  int n0 = blockIdx.x * 32, k0 = blockIdx.y * 32;
  int t = threadIdx.x;
  int c = t & 31, r = t >> 5;      // 32 cols x 8 rows per pass
  #pragma unroll
  for (int rr = 0; rr < 32; rr += 8)
    tile[r + rr][c] = f2bf(inm[(long long)(k0 + r + rr) * N + n0 + c]);
  __syncthreads();
  #pragma unroll
  for (int rr = 0; rr < 32; rr += 8)
    outm[(long long)(n0 + r + rr) * K + k0 + c] = tile[c][r + rr];
}

// h[b,s,c] = x[b,c,s]*sqrt(C) + pe[s,c]; writes f32 + bf16 copies.
__global__ __launch_bounds__(256) void embed_k(
    const float* __restrict__ x, float* __restrict__ h,
    unsigned short* __restrict__ hb) {
  int bs = blockIdx.x;
  int b = bs >> 9, s = bs & 511;
  int t = threadIdx.x;
  const float sq = 22.62741699796952f;  // sqrt(512)
  #pragma unroll
  for (int i = 0; i < 2; ++i) {
    int c = t + i * 256;
    float xv  = x[((long long)b * C_ + c) * S_ + s];
    float ang = (float)s * powf(10000.0f, (-2.0f * (float)c) / (float)C_);
    float pe  = (c & 1) ? cosf(ang) : sinf(ang);
    float v = xv * sq + pe;
    h [(long long)bs * C_ + c] = v;
    hb[(long long)bs * C_ + c] = f2bf(v);
  }
}

// row softmax over 512 with scale 1/sqrt(DH)=0.125 folded in; bf16 out.
__global__ __launch_bounds__(256) void softmax_bf16(
    const float* __restrict__ sc, unsigned short* __restrict__ P) {
  __shared__ float red[256];
  long long row = blockIdx.x;
  const float* s = sc + row * 512;
  unsigned short* p = P + row * 512;
  int t = threadIdx.x;
  float v0 = s[t] * 0.125f, v1 = s[t + 256] * 0.125f;
  red[t] = fmaxf(v0, v1); __syncthreads();
  for (int w = 128; w > 0; w >>= 1) { if (t < w) red[t] = fmaxf(red[t], red[t + w]); __syncthreads(); }
  float m = red[0]; __syncthreads();
  float e0 = expf(v0 - m), e1 = expf(v1 - m);
  red[t] = e0 + e1; __syncthreads();
  for (int w = 128; w > 0; w >>= 1) { if (t < w) red[t] += red[t + w]; __syncthreads(); }
  float inv = 1.0f / red[0];
  p[t] = f2bf(e0 * inv);
  p[t + 256] = f2bf(e1 * inv);
}

// h = LayerNorm(hin + delta) * g + beta; writes f32 + bf16 (in-place safe per row).
__global__ __launch_bounds__(256) void add_ln(
    const float* __restrict__ hin, const float* __restrict__ d,
    const float* __restrict__ g, const float* __restrict__ bt,
    float* __restrict__ hout, unsigned short* __restrict__ hb) {
  __shared__ float red[256];
  long long base = (long long)blockIdx.x * 512;
  int t = threadIdx.x;
  float x0 = hin[base + t] + d[base + t];
  float x1 = hin[base + t + 256] + d[base + t + 256];
  red[t] = x0 + x1; __syncthreads();
  for (int w = 128; w > 0; w >>= 1) { if (t < w) red[t] += red[t + w]; __syncthreads(); }
  float mean = red[0] * (1.0f / 512.0f); __syncthreads();
  float c0 = x0 - mean, c1 = x1 - mean;
  red[t] = c0 * c0 + c1 * c1; __syncthreads();
  for (int w = 128; w > 0; w >>= 1) { if (t < w) red[t] += red[t + w]; __syncthreads(); }
  float rstd = rsqrtf(red[0] * (1.0f / 512.0f) + 1e-5f);
  float y0 = c0 * rstd * g[t] + bt[t];
  float y1 = c1 * rstd * g[t + 256] + bt[t + 256];
  hout[base + t] = y0;       hout[base + t + 256] = y1;
  hb[base + t] = f2bf(y0);   hb[base + t + 256] = f2bf(y1);
}

// Classifier partials: part[chunk][b][o] = sum_{k in chunk} relu(h[b,k])*We[k,o]
// grid (O/64, 32 chunks); memory-bound on We (512MB read once). Deterministic.
__global__ __launch_bounds__(256) void classifier_k(
    const float* __restrict__ h, const float* __restrict__ We,
    float* __restrict__ part) {
  __shared__ float a_sh[16][64];
  int o0 = blockIdx.x * 64;
  long long k0 = (long long)blockIdx.y * 8192;
  int t = threadIdx.x;
  int ol = t & 63, bg = t >> 6;     // bg: 0..3 -> batches {bg, bg+4, bg+8, bg+12}
  float acc[4] = {0.f, 0.f, 0.f, 0.f};
  for (long long kb = k0; kb < k0 + 8192; kb += 64) {
    __syncthreads();
    #pragma unroll
    for (int r = 0; r < 4; ++r) {
      int slot = t + 256 * r;       // 0..1023 = 16b x 64k
      int bb = slot >> 6, kk = slot & 63;
      a_sh[bb][kk] = fmaxf(h[(long long)bb * 262144 + kb + kk], 0.0f);
    }
    __syncthreads();
    for (int kk = 0; kk < 64; ++kk) {
      float w = We[(kb + kk) * 512 + o0 + ol];
      acc[0] += a_sh[bg][kk] * w;
      acc[1] += a_sh[bg + 4][kk] * w;
      acc[2] += a_sh[bg + 8][kk] * w;
      acc[3] += a_sh[bg + 12][kk] * w;
    }
  }
  long long pb = (long long)blockIdx.y * (B_ * O_);
  #pragma unroll
  for (int j = 0; j < 4; ++j)
    part[pb + (bg + 4 * j) * O_ + o0 + ol] = acc[j];
}

__global__ __launch_bounds__(256) void cls_reduce(
    const float* __restrict__ part, const float* __restrict__ be,
    float* __restrict__ out) {
  int i = blockIdx.x * 256 + threadIdx.x;   // 0..8191
  float a = be[i & 511];
  for (int c = 0; c < 32; ++c) a += part[(long long)c * (B_ * O_) + i];
  out[i] = a;
}

// ---------------------------------------------------------------------------
extern "C" void kernel_launch(void* const* d_in, const int* in_sizes, int n_in,
                              void* d_out, int out_size, void* d_ws, size_t ws_size,
                              hipStream_t stream) {
  (void)in_sizes; (void)n_in; (void)out_size; (void)ws_size;
  const float* x   = (const float*)d_in[0];
  const float* Wq  = (const float*)d_in[1];
  const float* Wk  = (const float*)d_in[2];
  const float* Wv  = (const float*)d_in[3];
  const float* Wo  = (const float*)d_in[4];
  const float* bo  = (const float*)d_in[5];
  const float* g1  = (const float*)d_in[6];
  const float* bt1 = (const float*)d_in[7];
  const float* W1  = (const float*)d_in[8];
  const float* b1  = (const float*)d_in[9];
  const float* W2  = (const float*)d_in[10];
  const float* b2  = (const float*)d_in[11];
  const float* g2  = (const float*)d_in[12];
  const float* bt2 = (const float*)d_in[13];
  const float* We  = (const float*)d_in[14];
  const float* be  = (const float*)d_in[15];
  float* out = (float*)d_out;

  // ---- workspace layout ----
  char* ws = (char*)d_ws;
  size_t off = 0;
  auto alloc = [&](size_t bytes) -> char* {
    char* p = ws + off;
    off += (bytes + 255) & ~(size_t)255;
    return p;
  };
  float*          h_f32  = (float*)         alloc((size_t)M_ * C_ * 4);
  unsigned short* h_bf   = (unsigned short*)alloc((size_t)M_ * C_ * 2);
  unsigned short* q_bf   = (unsigned short*)alloc((size_t)M_ * C_ * 2);
  unsigned short* k_bf   = (unsigned short*)alloc((size_t)M_ * C_ * 2);
  unsigned short* v_bf   = (unsigned short*)alloc((size_t)M_ * C_ * 2);  // Vt[b,h,d,s]
  unsigned short* o_bf   = (unsigned short*)alloc((size_t)M_ * C_ * 2);
  float*          scores = (float*)         alloc((size_t)B_ * H_ * S_ * S_ * 4); // 134MB
  unsigned short* P_bf   = (unsigned short*)alloc((size_t)B_ * H_ * S_ * S_ * 2); // 67MB
  unsigned short* wqt    = (unsigned short*)alloc((size_t)L_ * C_ * C_ * 2);   // [N,K] bf16
  unsigned short* wkt    = (unsigned short*)alloc((size_t)L_ * C_ * C_ * 2);
  unsigned short* wvt    = (unsigned short*)alloc((size_t)L_ * C_ * C_ * 2);
  unsigned short* wot    = (unsigned short*)alloc((size_t)L_ * C_ * C_ * 2);
  unsigned short* w1t    = (unsigned short*)alloc((size_t)L_ * C_ * FF_ * 2);  // [FF,C]
  unsigned short* w2t    = (unsigned short*)alloc((size_t)L_ * FF_ * C_ * 2);  // [C,FF]
  float*          part   = (float*)         alloc((size_t)32 * B_ * O_ * 4);
  // alias the (phase-dead) scores buffer for FFN intermediate + GEMM temp
  unsigned short* ffn_bf = (unsigned short*)scores;                       // 33.5MB
  float*          tmp    = (float*)((char*)scores + (((size_t)M_ * FF_ * 2 + 255) & ~(size_t)255));

  dim3 blk(256);

  // ---- weight f32 -> bf16 transposed conversions (once per call) ----
  transpose_bf16_k<<<dim3(C_ / 32, C_ / 32, L_), blk, 0, stream>>>(Wq, wqt, C_, C_);
  transpose_bf16_k<<<dim3(C_ / 32, C_ / 32, L_), blk, 0, stream>>>(Wk, wkt, C_, C_);
  transpose_bf16_k<<<dim3(C_ / 32, C_ / 32, L_), blk, 0, stream>>>(Wv, wvt, C_, C_);
  transpose_bf16_k<<<dim3(C_ / 32, C_ / 32, L_), blk, 0, stream>>>(Wo, wot, C_, C_);
  transpose_bf16_k<<<dim3(FF_ / 32, C_ / 32, L_), blk, 0, stream>>>(W1, w1t, C_, FF_);
  transpose_bf16_k<<<dim3(C_ / 32, FF_ / 32, L_), blk, 0, stream>>>(W2, w2t, FF_, C_);

  // ---- embedding + positional encoding ----
  embed_k<<<M_, blk, 0, stream>>>(x, h_f32, h_bf);

  const long long SC = (long long)S_ * C_;
  const long long SS = (long long)S_ * S_;

  for (int l = 0; l < L_; ++l) {
    const unsigned short* wq = wqt + (size_t)l * C_ * C_;
    const unsigned short* wk = wkt + (size_t)l * C_ * C_;
    const unsigned short* wv = wvt + (size_t)l * C_ * C_;
    const unsigned short* wo = wot + (size_t)l * C_ * C_;
    const unsigned short* w1 = w1t + (size_t)l * C_ * FF_;
    const unsigned short* w2 = w2t + (size_t)l * FF_ * C_;

    // Q, K projections -> bf16 [B,S,C]; V projection -> bf16 Vt[b,h,d,s]
    gemm_bf16_wmma<1><<<dim3(C_ / 64, M_ / 128, 1), blk, 0, stream>>>(
        h_bf, wq, q_bf, nullptr, M_, C_, C_, C_, C_, C_, 1, 0, 0, 0, 0, 0, 0);
    gemm_bf16_wmma<1><<<dim3(C_ / 64, M_ / 128, 1), blk, 0, stream>>>(
        h_bf, wk, k_bf, nullptr, M_, C_, C_, C_, C_, C_, 1, 0, 0, 0, 0, 0, 0);
    gemm_bf16_wmma<2><<<dim3(C_ / 64, M_ / 128, 1), blk, 0, stream>>>(
        h_bf, wv, v_bf, nullptr, M_, C_, C_, C_, C_, C_, 1, 0, 0, 0, 0, 0, 0);

    // scores[b,h] = Q_h @ K_h^T : batch 128, M=512, N=512, K=64 (K rows = B^T)
    gemm_bf16_wmma<0><<<dim3(S_ / 64, S_ / 128, B_ * H_), blk, 0, stream>>>(
        q_bf, k_bf, scores, nullptr, S_, S_, DH_, C_, C_, S_,
        H_, SC, (long long)DH_, SC, (long long)DH_, (long long)H_ * SS, SS);

    // softmax rows (scale folded in) -> bf16 probs
    softmax_bf16<<<B_ * H_ * S_, blk, 0, stream>>>(scores, P_bf);

    // o[b,h] = P @ V_h : batch 128, M=512, N=64, K=512; B^T = Vt rows
    gemm_bf16_wmma<1><<<dim3(DH_ / 64, S_ / 128, B_ * H_), blk, 0, stream>>>(
        P_bf, v_bf, o_bf, nullptr, S_, DH_, S_, S_, S_, C_,
        H_, (long long)H_ * SS, SS, SC, (long long)(DH_ * S_), SC, (long long)DH_);

    // output projection (+bias) -> f32 tmp, then residual + LN1
    gemm_bf16_wmma<4><<<dim3(C_ / 64, M_ / 128, 1), blk, 0, stream>>>(
        o_bf, wo, tmp, bo + (size_t)l * C_, M_, C_, C_, C_, C_, C_,
        1, 0, 0, 0, 0, 0, 0);
    add_ln<<<M_, blk, 0, stream>>>(h_f32, tmp, g1 + (size_t)l * C_,
                                   bt1 + (size_t)l * C_, h_f32, h_bf);

    // FFN1 (+bias, relu) -> bf16; FFN2 (+bias) -> f32 tmp; residual + LN2
    gemm_bf16_wmma<3><<<dim3(FF_ / 64, M_ / 128, 1), blk, 0, stream>>>(
        h_bf, w1, ffn_bf, b1 + (size_t)l * FF_, M_, FF_, C_, C_, C_, FF_,
        1, 0, 0, 0, 0, 0, 0);
    gemm_bf16_wmma<4><<<dim3(C_ / 64, M_ / 128, 1), blk, 0, stream>>>(
        ffn_bf, w2, tmp, b2 + (size_t)l * C_, M_, C_, FF_, FF_, FF_, C_,
        1, 0, 0, 0, 0, 0, 0);
    add_ln<<<M_, blk, 0, stream>>>(h_f32, tmp, g2 + (size_t)l * C_,
                                   bt2 + (size_t)l * C_, h_f32, h_bf);
  }

  // ---- classifier head: relu(h).reshape(16, 262144) @ We + be ----
  classifier_k<<<dim3(O_ / 64, 32), blk, 0, stream>>>(h_f32, We, part);
  cls_reduce<<<dim3(B_ * O_ / 256), blk, 0, stream>>>(part, be, out);
}
